// StageAProcessor_60017872994702
// MI455X (gfx1250) — matrix operand order repaired and checked by reading, not compiled, not run on hardware
//
#include <hip/hip_runtime.h>
#include <hip/hip_bf16.h>

// ---------------------------------------------------------------------------
// Joint txt+img attention block for MI455X (gfx1250, wave32, WMMA).
//
// Shapes: SEQ_TXT=256, L_IMG=4096, L=4352, HEADS=16, HEAD_DIM=128, D=2048.
// Mask regions (all boundaries multiples of 32):
//   q in [0,2304)    (txt+syn): attends [0,4352)
//   q in [2304,3328) (ref)    : attends [2304,3328)
//   q in [3328,4352) (src)    : attends [2304,4352)
// ---------------------------------------------------------------------------

#define LFULL   4352
#define SEQT    256
#define LIMG    4096
#define DMODEL  2048
#define NHEADS  16
#define HDIM    128

typedef __attribute__((ext_vector_type(4)))  __bf16 bf16x4;
typedef __attribute__((ext_vector_type(8)))  __bf16 bf16x8;
typedef __attribute__((ext_vector_type(16))) __bf16 bf16x16;
typedef __attribute__((ext_vector_type(8)))  float  f32x8;
typedef __attribute__((ext_vector_type(4)))  float  f32x4;

// XOR-lane shuffle via ds_swizzle (group-of-32 mode: and=0x1f, or=0, xor=M).
// No index VGPR, no LDS memory traffic.
template <int M>
__device__ inline float xor_swizzle(float v) {
  int r = __builtin_amdgcn_ds_swizzle(__builtin_bit_cast(int, v),
                                      0x1f | (M << 10));
  return __builtin_bit_cast(float, r);
}

// A-fragment (16x32 bf16): lane holds K = half*16 + [0..15], contiguous
__device__ inline bf16x16 ld_a_frag(const __bf16* p) {
  bf16x8 lo = *(const bf16x8*)p;
  bf16x8 hi = *(const bf16x8*)(p + 8);
  return __builtin_shufflevector(lo, hi, 0,1,2,3,4,5,6,7,8,9,10,11,12,13,14,15);
}
// B-fragment (32x16 bf16): lane (col) holds K = half*16 + [0..15], contiguous
__device__ inline bf16x16 ld_b_frag(const __bf16* p) {
  bf16x8 lo = *(const bf16x8*)p;
  bf16x8 hi = *(const bf16x8*)(p + 8);
  return __builtin_shufflevector(lo, hi, 0,1,2,3,4,5,6,7,8,9,10,11,12,13,14,15);
}

__device__ inline f32x8 wmma_bf16(bf16x16 a, bf16x16 b, f32x8 c) {
  return __builtin_amdgcn_wmma_f32_16x16x32_bf16(
      /*neg_a=*/false, a, /*neg_b=*/false, b,
      /*c_mod=*/(short)0, c, /*reuse_a=*/false, /*reuse_b=*/false);
}

// ---------------------------------------------------------------------------
// K0a: activations f32 -> bf16.  Row layout: rows [0,256)=txt, [256,4352)=img.
// ---------------------------------------------------------------------------
__global__ void convert_x_kernel(const float* __restrict__ hidden,
                                 const float* __restrict__ enc,
                                 __bf16* __restrict__ Xall) {
  size_t idx = ((size_t)blockIdx.x * blockDim.x + threadIdx.x) * 4;
  int row = (int)(idx >> 11);
  int col = (int)(idx & 2047);
  const float* src = (row < SEQT) ? (enc + (size_t)row * DMODEL + col)
                                  : (hidden + (size_t)(row - SEQT) * DMODEL + col);
  f32x4 v = *(const f32x4*)src;
  bf16x4 o;
  o.x = (__bf16)v.x; o.y = (__bf16)v.y; o.z = (__bf16)v.z; o.w = (__bf16)v.w;
  *(bf16x4*)(Xall + idx) = o;
}

// ---------------------------------------------------------------------------
// K0b: weight transpose-convert: Wt[n][k] = (bf16)W[k][n]   (2048x2048)
// ---------------------------------------------------------------------------
__global__ void wtrans_kernel(const float* __restrict__ W, __bf16* __restrict__ Wt) {
  __shared__ float tile[32][33];
  int n0 = blockIdx.x * 32, k0 = blockIdx.y * 32;
  int tx = threadIdx.x, ty = threadIdx.y;            // block (32,8)
  #pragma unroll
  for (int r = ty; r < 32; r += 8)
    tile[r][tx] = W[(size_t)(k0 + r) * DMODEL + n0 + tx];
  __syncthreads();
  #pragma unroll
  for (int r = ty; r < 32; r += 8)
    Wt[(size_t)(n0 + r) * DMODEL + k0 + tx] = (__bf16)tile[tx][r];
}

// ---------------------------------------------------------------------------
// K1: C[M,2048] = A[M,2048] @ Wt^T + bias.   A,Wt bf16, f32 accumulate.
// One wave owns a 16x64 output tile (4 WMMAs per K-step, A-frag reused).
// Block = 256 threads = 8 waves -> 512 output columns; grid (4, M/16).
// ---------------------------------------------------------------------------
template <typename OutT>
__global__ void __launch_bounds__(256, 1)
gemm_nt_bias_kernel(const __bf16* __restrict__ A,
                    const __bf16* __restrict__ Wt,
                    const float* __restrict__ bias,
                    OutT* __restrict__ C, int M) {
  const int wave = threadIdx.x >> 5;
  const int lane = threadIdx.x & 31;
  const int half = lane >> 4;
  const int lr   = lane & 15;
  const int m0 = blockIdx.y * 16;
  const int n0 = (blockIdx.x * 8 + wave) * 64;

  f32x8 acc0 = {0,0,0,0,0,0,0,0}, acc1 = acc0, acc2 = acc0, acc3 = acc0;

  const __bf16* Arow = A + (size_t)(m0 + lr) * DMODEL + half * 16;
  const __bf16* B0 = Wt + (size_t)(n0 +  0 + lr) * DMODEL + half * 16;
  const __bf16* B1 = Wt + (size_t)(n0 + 16 + lr) * DMODEL + half * 16;
  const __bf16* B2 = Wt + (size_t)(n0 + 32 + lr) * DMODEL + half * 16;
  const __bf16* B3 = Wt + (size_t)(n0 + 48 + lr) * DMODEL + half * 16;

  for (int k = 0; k < DMODEL; k += 32) {
    __builtin_prefetch(Arow + k + 256, 0, 1);      // global_prefetch_b8
    bf16x16 af = ld_a_frag(Arow + k);
    acc0 = wmma_bf16(af, ld_b_frag(B0 + k), acc0);
    acc1 = wmma_bf16(af, ld_b_frag(B1 + k), acc1);
    acc2 = wmma_bf16(af, ld_b_frag(B2 + k), acc2);
    acc3 = wmma_bf16(af, ld_b_frag(B3 + k), acc3);
  }

  f32x8 accs[4] = {acc0, acc1, acc2, acc3};
  #pragma unroll
  for (int t = 0; t < 4; ++t) {
    int n = n0 + t * 16 + lr;
    float bv = bias[n];
    #pragma unroll
    for (int j = 0; j < 8; ++j) {
      int m = m0 + j + half * 8;                   // C-layout row mapping
      C[(size_t)m * DMODEL + n] = (OutT)(accs[t][j] + bv);
    }
  }
}

// ---------------------------------------------------------------------------
// K2: per-(pos,head) RMSNorm + RoPE on Q and K.  One wave per row of 128.
// Writes Q,K as [head][pos][128] bf16 (A-side layout for score WMMAs).
// The 1/sqrt(HEAD_DIM) attention scale is folded into Q here (linear op,
// commutes with the QK^T product), so the flash kernel skips it entirely.
// ---------------------------------------------------------------------------
__global__ void qk_postprocess_kernel(const __bf16* __restrict__ Qraw,
                                      const __bf16* __restrict__ Kraw,
                                      const float* __restrict__ gq,
                                      const float* __restrict__ gk,
                                      const float* __restrict__ gaq,
                                      const float* __restrict__ gak,
                                      const float* __restrict__ img_cos,
                                      const float* __restrict__ img_sin,
                                      const float* __restrict__ txt_cos,
                                      const float* __restrict__ txt_sin,
                                      __bf16* __restrict__ Qout,
                                      __bf16* __restrict__ Kout) {
  const int wave = threadIdx.x >> 5;
  const int lane = threadIdx.x & 31;
  const int idx = blockIdx.x * 8 + wave;           // 0 .. L*H-1
  const int pos = idx >> 4;
  const int h   = idx & 15;
  const bool is_txt = pos < SEQT;
  const float qscale = 0.08838834764831845f;       // 1/sqrt(128)

  const float* gQ = is_txt ? gaq : gq;
  const float* gK = is_txt ? gak : gk;
  const float* crow = is_txt ? (txt_cos + (size_t)pos * 64)
                             : (img_cos + (size_t)(pos - SEQT) * 64);
  const float* srow = is_txt ? (txt_sin + (size_t)pos * 64)
                             : (img_sin + (size_t)(pos - SEQT) * 64);

  const size_t in_off  = (size_t)pos * DMODEL + h * HDIM + lane * 4;
  const size_t out_off = ((size_t)h * LFULL + pos) * HDIM + lane * 4;

  float c0 = crow[lane * 2], s0 = srow[lane * 2];
  float c1 = crow[lane * 2 + 1], s1 = srow[lane * 2 + 1];
  float g0 = gQ[lane * 4], g1 = gQ[lane * 4 + 1], g2 = gQ[lane * 4 + 2], g3 = gQ[lane * 4 + 3];
  float h0 = gK[lane * 4], h1 = gK[lane * 4 + 1], h2 = gK[lane * 4 + 2], h3 = gK[lane * 4 + 3];

  // ---- Q (carries the attention scale) ----
  {
    bf16x4 r = *(const bf16x4*)(Qraw + in_off);
    float x0 = (float)r.x, x1 = (float)r.y, x2 = (float)r.z, x3 = (float)r.w;
    float ss = x0*x0 + x1*x1 + x2*x2 + x3*x3;
    ss += xor_swizzle<1>(ss);  ss += xor_swizzle<2>(ss);
    ss += xor_swizzle<4>(ss);  ss += xor_swizzle<8>(ss);
    ss += xor_swizzle<16>(ss);
    float rms = rsqrtf(ss * (1.0f / 128.0f) + 1e-6f);
    x0 *= rms * g0; x1 *= rms * g1; x2 *= rms * g2; x3 *= rms * g3;
    bf16x4 o;
    o.x = (__bf16)((x0 * c0 - x1 * s0) * qscale);
    o.y = (__bf16)((x0 * s0 + x1 * c0) * qscale);
    o.z = (__bf16)((x2 * c1 - x3 * s1) * qscale);
    o.w = (__bf16)((x2 * s1 + x3 * c1) * qscale);
    *(bf16x4*)(Qout + out_off) = o;
  }
  // ---- K ----
  {
    bf16x4 r = *(const bf16x4*)(Kraw + in_off);
    float x0 = (float)r.x, x1 = (float)r.y, x2 = (float)r.z, x3 = (float)r.w;
    float ss = x0*x0 + x1*x1 + x2*x2 + x3*x3;
    ss += xor_swizzle<1>(ss);  ss += xor_swizzle<2>(ss);
    ss += xor_swizzle<4>(ss);  ss += xor_swizzle<8>(ss);
    ss += xor_swizzle<16>(ss);
    float rms = rsqrtf(ss * (1.0f / 128.0f) + 1e-6f);
    x0 *= rms * h0; x1 *= rms * h1; x2 *= rms * h2; x3 *= rms * h3;
    bf16x4 o;
    o.x = (__bf16)(x0 * c0 - x1 * s0);  o.y = (__bf16)(x0 * s0 + x1 * c0);
    o.z = (__bf16)(x2 * c1 - x3 * s1);  o.w = (__bf16)(x2 * s1 + x3 * c1);
    *(bf16x4*)(Kout + out_off) = o;
  }
}

// ---------------------------------------------------------------------------
// K2b: V transpose per head via LDS: Vt[h][d][pos] = Vraw[pos][h*128+d]
// (so PV B-fragments are contiguous along the kv dimension).
// ---------------------------------------------------------------------------
__global__ void v_transpose_kernel(const __bf16* __restrict__ Vraw,
                                   __bf16* __restrict__ Vt) {
  __shared__ __bf16 tile[32][33];
  const int h = blockIdx.z;
  const int d0 = blockIdx.x * 32, l0 = blockIdx.y * 32;
  const int tx = threadIdx.x, ty = threadIdx.y;    // block (32,8)
  #pragma unroll
  for (int r = ty; r < 32; r += 8)
    tile[r][tx] = Vraw[(size_t)(l0 + r) * DMODEL + h * HDIM + d0 + tx];
  __syncthreads();
  #pragma unroll
  for (int r = ty; r < 32; r += 8)
    Vt[((size_t)h * HDIM + d0 + r) * LFULL + l0 + tx] = tile[tx][r];
}

// ---------------------------------------------------------------------------
// K3: flash attention.  One wave = one 16-row query tile of one head.
// kv step 32: 8 score WMMAs -> online softmax (ds_swizzle row-max; row-sum
// computed with a P @ ones WMMA) -> P via LDS -> 8 PV WMMAs.
// Masked key blocks are skipped outright (block mask is 32-aligned).
// ---------------------------------------------------------------------------
__global__ void __launch_bounds__(256, 1)
flash_attn_kernel(const __bf16* __restrict__ Q,
                  const __bf16* __restrict__ Kmat,
                  const __bf16* __restrict__ Vt,
                  __bf16* __restrict__ Aout) {
  __shared__ __bf16 plds[8][16][32];               // per-wave P staging (8 KB)
  const int wave = threadIdx.x >> 5;
  const int lane = threadIdx.x & 31;
  const int half = lane >> 4;
  const int lr   = lane & 15;
  const int h  = blockIdx.y;
  const int q0 = (blockIdx.x * 8 + wave) * 16;

  int kv_lo, kv_hi;
  if (q0 < SEQT + 2048)            { kv_lo = 0;    kv_hi = LFULL; }   // txt+syn
  else if (q0 < SEQT + 2048 + 1024){ kv_lo = 2304; kv_hi = 3328;  }   // ref
  else                             { kv_lo = 2304; kv_hi = LFULL; }   // src

  const __bf16* Qh = Q    + ((size_t)h * LFULL + q0) * HDIM;
  const __bf16* Kh = Kmat + (size_t)h * LFULL * HDIM;
  const __bf16* Vh = Vt   + (size_t)h * HDIM * LFULL;

  bf16x16 qf[4];
  #pragma unroll
  for (int c = 0; c < 4; ++c)
    qf[c] = ld_a_frag(Qh + (size_t)lr * HDIM + c * 32 + half * 16);

  bf16x16 onesB;
  #pragma unroll
  for (int e = 0; e < 16; ++e) onesB[e] = (__bf16)1.0f;

  f32x8 o[8];
  #pragma unroll
  for (int t = 0; t < 8; ++t) o[t] = (f32x8){0,0,0,0,0,0,0,0};
  float mrow[8], lrow[8];
  #pragma unroll
  for (int j = 0; j < 8; ++j) { mrow[j] = -3.0e38f; lrow[j] = 0.0f; }

  for (int kv = kv_lo; kv < kv_hi; kv += 32) {
    f32x8 s0 = (f32x8){0,0,0,0,0,0,0,0};
    f32x8 s1 = (f32x8){0,0,0,0,0,0,0,0};
    #pragma unroll
    for (int c = 0; c < 4; ++c) {
      s0 = wmma_bf16(qf[c], ld_b_frag(Kh + (size_t)(kv      + lr) * HDIM + c * 32 + half * 16), s0);
      s1 = wmma_bf16(qf[c], ld_b_frag(Kh + (size_t)(kv + 16 + lr) * HDIM + c * 32 + half * 16), s1);
    }

    float alpha[8];
    #pragma unroll
    for (int j = 0; j < 8; ++j) {
      float a = s0[j], b = s1[j];                  // scale already folded into Q
      float mx = fmaxf(a, b);
      mx = fmaxf(mx, xor_swizzle<1>(mx));
      mx = fmaxf(mx, xor_swizzle<2>(mx));
      mx = fmaxf(mx, xor_swizzle<4>(mx));
      mx = fmaxf(mx, xor_swizzle<8>(mx));          // row max over 32 cols
      float mn = fmaxf(mrow[j], mx);
      alpha[j] = __expf(mrow[j] - mn);
      mrow[j] = mn;
      const int rr = j + half * 8;                 // C-layout row
      plds[wave][rr][lr]      = (__bf16)__expf(a - mn);
      plds[wave][rr][lr + 16] = (__bf16)__expf(b - mn);
    }
    #pragma unroll
    for (int t = 0; t < 8; ++t)
      #pragma unroll
      for (int j = 0; j < 8; ++j)
        o[t][j] = o[t][j] * alpha[j];

    // wave-private LDS RAW fence (DS ops are in-order per wave)
    asm volatile("s_wait_dscnt 0" ::: "memory");
    bf16x16 pf = ld_a_frag(&plds[wave][lr][half * 16]);

    // row sums of P via matrix pipe: every column of P @ ones is rowsum(P)
    f32x8 ls = wmma_bf16(pf, onesB, (f32x8){0,0,0,0,0,0,0,0});
    #pragma unroll
    for (int j = 0; j < 8; ++j)
      lrow[j] = lrow[j] * alpha[j] + ls[j];

    #pragma unroll
    for (int t = 0; t < 8; ++t)
      o[t] = wmma_bf16(pf, ld_b_frag(Vh + (size_t)(t * 16 + lr) * LFULL + kv + half * 16), o[t]);
  }

  float inv[8];
  #pragma unroll
  for (int j = 0; j < 8; ++j) inv[j] = 1.0f / lrow[j];
  #pragma unroll
  for (int t = 0; t < 8; ++t)
    #pragma unroll
    for (int j = 0; j < 8; ++j) {
      int m = q0 + j + half * 8;
      Aout[(size_t)m * DMODEL + h * HDIM + t * 16 + lr] = (__bf16)(o[t][j] * inv[j]);
    }
}

// ---------------------------------------------------------------------------
// Host-side orchestration
// ---------------------------------------------------------------------------
extern "C" void kernel_launch(void* const* d_in, const int* in_sizes, int n_in,
                              void* d_out, int out_size, void* d_ws, size_t ws_size,
                              hipStream_t stream) {
  const float* hidden  = (const float*)d_in[0];
  const float* enc     = (const float*)d_in[1];
  const float* wq  = (const float*)d_in[2];   const float* bq  = (const float*)d_in[3];
  const float* wk  = (const float*)d_in[4];   const float* bk  = (const float*)d_in[5];
  const float* wv  = (const float*)d_in[6];   const float* bv  = (const float*)d_in[7];
  const float* awq = (const float*)d_in[8];   const float* abq = (const float*)d_in[9];
  const float* awk = (const float*)d_in[10];  const float* abk = (const float*)d_in[11];
  const float* awv = (const float*)d_in[12];  const float* abv = (const float*)d_in[13];
  const float* wo  = (const float*)d_in[14];  const float* bo  = (const float*)d_in[15];
  const float* awo = (const float*)d_in[16];  const float* abo = (const float*)d_in[17];
  const float* gq  = (const float*)d_in[18];  const float* gk  = (const float*)d_in[19];
  const float* gaq = (const float*)d_in[20];  const float* gak = (const float*)d_in[21];
  const float* img_cos = (const float*)d_in[22];
  const float* img_sin = (const float*)d_in[23];
  const float* txt_cos = (const float*)d_in[24];
  const float* txt_sin = (const float*)d_in[25];
  float* out = (float*)d_out;

  // ---- workspace carve-up ----
  char* ws = (char*)d_ws;
  const size_t ACT  = (size_t)LFULL * DMODEL * sizeof(__bf16);   // 17,825,792
  const size_t WMAT = (size_t)DMODEL * DMODEL * sizeof(__bf16);  //  8,388,608
  __bf16* Xall = (__bf16*)ws;                   ws += ACT;
  __bf16* Wt[8];
  for (int i = 0; i < 8; ++i) { Wt[i] = (__bf16*)ws; ws += WMAT; }
  __bf16* Qraw = (__bf16*)ws;                   ws += ACT;
  __bf16* Kraw = (__bf16*)ws;                   ws += ACT;
  __bf16* Vraw = (__bf16*)ws;                   ws += ACT;
  __bf16* Qbf  = (__bf16*)ws;                   ws += ACT;
  __bf16* Kbf  = (__bf16*)ws;                   ws += ACT;
  __bf16* Vtbf = (__bf16*)ws;                   ws += ACT;
  __bf16* attn = Qraw;   // alias: Qraw is dead after qk_postprocess

  // 1) activations -> bf16
  convert_x_kernel<<<dim3(8704), dim3(256), 0, stream>>>(hidden, enc, Xall);

  // 2) weights -> transposed bf16
  const float* Wsrc[8] = {wq, wk, wv, awq, awk, awv, wo, awo};
  for (int i = 0; i < 8; ++i)
    wtrans_kernel<<<dim3(64, 64), dim3(32, 8), 0, stream>>>(Wsrc[i], Wt[i]);

  // 3) QKV projections (txt rows [0,256) use aw*, img rows [256,4352) use w*)
  const dim3 gImg(4, LIMG / 16), gTxt(4, SEQT / 16), blk(256);
  gemm_nt_bias_kernel<__bf16><<<gTxt, blk, 0, stream>>>(Xall,                 Wt[3], abq, Qraw,                 SEQT);
  gemm_nt_bias_kernel<__bf16><<<gImg, blk, 0, stream>>>(Xall + SEQT * DMODEL, Wt[0], bq,  Qraw + SEQT * DMODEL, LIMG);
  gemm_nt_bias_kernel<__bf16><<<gTxt, blk, 0, stream>>>(Xall,                 Wt[4], abk, Kraw,                 SEQT);
  gemm_nt_bias_kernel<__bf16><<<gImg, blk, 0, stream>>>(Xall + SEQT * DMODEL, Wt[1], bk,  Kraw + SEQT * DMODEL, LIMG);
  gemm_nt_bias_kernel<__bf16><<<gTxt, blk, 0, stream>>>(Xall,                 Wt[5], abv, Vraw,                 SEQT);
  gemm_nt_bias_kernel<__bf16><<<gImg, blk, 0, stream>>>(Xall + SEQT * DMODEL, Wt[2], bv,  Vraw + SEQT * DMODEL, LIMG);

  // 4) RMSNorm + RoPE on Q/K (Q also gets 1/sqrt(d)); V transpose per head
  qk_postprocess_kernel<<<dim3(8704), blk, 0, stream>>>(
      Qraw, Kraw, gq, gk, gaq, gak, img_cos, img_sin, txt_cos, txt_sin, Qbf, Kbf);
  v_transpose_kernel<<<dim3(4, 136, 16), dim3(32, 8), 0, stream>>>(Vraw, Vtbf);

  // 5) flash attention (272 query tiles x 16 heads; 8 tiles per block)
  flash_attn_kernel<<<dim3(34, 16), blk, 0, stream>>>(Qbf, Kbf, Vtbf, attn);

  // 6) output projections: img_out first in d_out, then txt_out
  gemm_nt_bias_kernel<float><<<gImg, blk, 0, stream>>>(attn + SEQT * DMODEL, Wt[6], bo,
                                                       out, LIMG);
  gemm_nt_bias_kernel<float><<<gTxt, blk, 0, stream>>>(attn, Wt[7], abo,
                                                       out + (size_t)LIMG * DMODEL, SEQT);
}
